// HigherOrderScore_65103114273470
// MI455X (gfx1250) — compile-verified, matrix operand order checked by reading, not run on hardware
//
#include <hip/hip_runtime.h>
#include <hip/hip_fp16.h>
#include <stdint.h>

#define NEGF (-1e10f)

typedef float    v2f  __attribute__((ext_vector_type(2)));
typedef float    v8f  __attribute__((ext_vector_type(8)));
typedef _Float16 h8   __attribute__((ext_vector_type(8)));
typedef _Float16 v16h __attribute__((ext_vector_type(16)));
typedef unsigned int u32x4 __attribute__((ext_vector_type(4)));
typedef int          i32x4 __attribute__((ext_vector_type(4)));
typedef int          i32x8 __attribute__((ext_vector_type(8)));

// ---------------------------------------------------------------------------
// Weight conversion: W1 [1596,1000] f32 -> W1t [1024,1600] f16 transposed,
// zero padded (k contiguous so TDM can DMA [128n x 32k] tiles straight to LDS)
// ---------------------------------------------------------------------------
__global__ void cvt_w1_kernel(const float* __restrict__ W1, _Float16* __restrict__ W1t) {
    int idx = blockIdx.x * 256 + threadIdx.x;          // 1024*1600
    if (idx >= 1024 * 1600) return;
    int n = idx / 1600, k = idx % 1600;
    float v = (k < 1596 && n < 1000) ? W1[k * 1000 + n] : 0.0f;
    W1t[idx] = (_Float16)v;
}

__global__ void cvt_aux_kernel(const float* __restrict__ b1, const float* __restrict__ W2,
                               float* __restrict__ b1p, float* __restrict__ W2p) {
    int n = blockIdx.x * 256 + threadIdx.x;
    if (n >= 1024) return;
    b1p[n] = (n < 1000) ? b1[n] : 0.0f;
    W2p[n] = (n < 1000) ? W2[n] : 0.0f;                // W2 is [1000,1]
}

// ---------------------------------------------------------------------------
// Coarse GEMM 1: T[1024,512] = g[1024,512] @ Wc^T   (fp32 WMMA 16x16x4)
// ---------------------------------------------------------------------------
__global__ void coarse1_kernel(const float* __restrict__ g, const float* __restrict__ Wc,
                               float* __restrict__ T) {
    int mB = blockIdx.x * 16, nB = blockIdx.y * 16;
    int lane = threadIdx.x;
    int l15 = lane & 15, hi = lane >> 4;
    const float* arow = g  + (size_t)(mB + l15) * 512;
    const float* brow = Wc + (size_t)(nB + l15) * 512;   // B[k][n] = Wc[n][k]
    v8f acc = {};
    for (int kb = 0; kb < 512; kb += 4) {
        int k0 = kb + (hi ? 2 : 0);
        v2f a; a.x = arow[k0]; a.y = arow[k0 + 1];
        v2f b; b.x = brow[k0]; b.y = brow[k0 + 1];
        acc = __builtin_amdgcn_wmma_f32_16x16x4_f32(false, a, false, b, (short)0, acc, false, false);
    }
    for (int r = 0; r < 8; ++r)
        T[(size_t)(mB + r + hi * 8) * 512 + nB + l15] = acc[r];
}

// ---------------------------------------------------------------------------
// Coarse GEMM 2: ant[1024,1024] = T @ g^T + s_i + s_j + mask  (fp32 WMMA)
// ---------------------------------------------------------------------------
__global__ void coarse2_kernel(const float* __restrict__ T, const float* __restrict__ g,
                               const float* __restrict__ ms, float* __restrict__ ant) {
    int mB = blockIdx.x * 16, nB = blockIdx.y * 16;
    int lane = threadIdx.x;
    int l15 = lane & 15, hi = lane >> 4;
    const float* arow = T + (size_t)(mB + l15) * 512;
    const float* brow = g + (size_t)(nB + l15) * 512;    // B[k][n] = g[n][k]
    v8f acc = {};
    for (int kb = 0; kb < 512; kb += 4) {
        int k0 = kb + (hi ? 2 : 0);
        v2f a; a.x = arow[k0]; a.y = arow[k0 + 1];
        v2f b; b.x = brow[k0]; b.y = brow[k0 + 1];
        acc = __builtin_amdgcn_wmma_f32_16x16x4_f32(false, a, false, b, (short)0, acc, false, false);
    }
    for (int r = 0; r < 8; ++r) {
        int m = mB + r + hi * 8;
        int n = nB + l15;
        float v = acc[r] + ms[m] + ms[n] + (((m - n) >= 1) ? 0.0f : NEGF);
        ant[(size_t)m * 1024 + n] = v;
    }
}

// ---------------------------------------------------------------------------
// Per-row top-50 (descending, ties -> lowest index), one 256-thread WG / row
// ---------------------------------------------------------------------------
__global__ void topk_kernel(const float* __restrict__ ant,
                            float* __restrict__ bestS, int* __restrict__ bestI) {
    __shared__ float buf[1024];
    __shared__ float rv[256];
    __shared__ int   ri[256];
    int row = blockIdx.x;
    int t = threadIdx.x;
    for (int c = t; c < 1024; c += 256) buf[c] = ant[(size_t)row * 1024 + c];
    __syncthreads();
    for (int sel = 0; sel < 50; ++sel) {
        float bv = -3.0e38f; int bi = 0;
        for (int c = t; c < 1024; c += 256) {
            float v = buf[c];
            if (v > bv || (v == bv && c < bi)) { bv = v; bi = c; }
        }
        rv[t] = bv; ri[t] = bi;
        __syncthreads();
        for (int s = 128; s > 0; s >>= 1) {
            if (t < s) {
                float v2 = rv[t + s]; int i2 = ri[t + s];
                if (v2 > rv[t] || (v2 == rv[t] && i2 < ri[t])) { rv[t] = v2; ri[t] = i2; }
            }
            __syncthreads();
        }
        if (t == 0) {
            bestS[row * 50 + sel] = rv[0];
            bestI[row * 50 + sel] = ri[0];
            buf[ri[0]] = -3.0e38f;
        }
        __syncthreads();
    }
}

// ---------------------------------------------------------------------------
// Pair features phi [51200, 64] f16 (60 real + 4 zero pad)
// ---------------------------------------------------------------------------
__global__ void phi_kernel(const int* __restrict__ startI, const int* __restrict__ endI,
                           const int* __restrict__ genreI, const int* __restrict__ spkI,
                           const float* __restrict__ distE, const float* __restrict__ genreE,
                           const float* __restrict__ spkE, const int* __restrict__ bestI,
                           _Float16* __restrict__ phi) {
    int idx = blockIdx.x * 256 + threadIdx.x;          // 51200*64
    if (idx >= 51200 * 64) return;
    int p = idx >> 6, f = idx & 63;
    int i = p / 50;
    int j = bestI[p];
    float v = 0.0f;
    if (f < 20) {
        int d = endI[i] - startI[j];
        int bin = (d > 1) + (d > 2) + (d > 3) + (d > 4) + (d > 8) + (d > 16) + (d > 32) + (d > 64);
        v = distE[bin * 20 + f];
    } else if (f < 40) {
        v = genreE[genreI[i] * 20 + (f - 20)];
    } else if (f < 60) {
        int lab = (spkI[i] == spkI[j]) ? 1 : 2;
        v = spkE[lab * 20 + (f - 40)];
    }
    phi[idx] = (_Float16)v;
}

// ---------------------------------------------------------------------------
// Score MLP: s[m] += sum_n relu( pairs[m,:]@W1[:,n] + b1[n] ) * W2[n]
// pairs built on the fly in LDS: [g_i | g_j | g_i*g_j | phi | 0pad], K=1600
// block tile 128x128, 8 waves, wave = 4x2 16x16 WMMA f16 tiles, K-step 32.
// B tile [128n x 32k] is DMA'd from W1t into LDS by the Tensor Data Mover
// (issued by wave 0, overlapped with the VALU A-tile build, s_wait_tensorcnt).
// ---------------------------------------------------------------------------
__global__ void __launch_bounds__(256)
mlp_gemm_kernel(const float* __restrict__ g, const _Float16* __restrict__ phi,
                const int* __restrict__ bestI, const _Float16* __restrict__ W1t,
                const float* __restrict__ b1p, const float* __restrict__ W2p,
                float* __restrict__ s_out) {
    __shared__ _Float16 As[128][32];
    __shared__ _Float16 Bt[128][32];
    int t = threadIdx.x;
    int mBase = blockIdx.x * 128;      // 400 groups, 51200 rows exactly
    int nBase = blockIdx.y * 128;      // 8 groups, 1024 padded cols
    int wave = t >> 5, lane = t & 31;
    int l15 = lane & 15, hi = lane >> 4;
    int wm = wave >> 2;                // 0..1 -> 64-row span
    int wn = wave & 3;                 // 0..3 -> 32-col span
    v8f acc[4][2] = {};

    unsigned int ldsB = (unsigned int)(uintptr_t)(&Bt[0][0]);   // LDS byte offset

    for (int kB = 0; kB < 1600; kB += 32) {
        // ---- wave 0: launch TDM DMA of the B tile: W1t[nBase..+128][kB..+32] ----
        if (wave == 0) {
            unsigned long long ga =
                (unsigned long long)(uintptr_t)(W1t + ((size_t)nBase * 1600 + kB));
            u32x4 g0;
            g0[0] = 1u;                                        // count = 1 (valid D#)
            g0[1] = ldsB;                                      // lds_addr
            g0[2] = (unsigned int)(ga & 0xFFFFFFFFu);          // global_addr[31:0]
            g0[3] = (unsigned int)((ga >> 32) & 0x01FFFFFFu)   // global_addr[56:32]
                    | (2u << 30);                              // type = 2 (image)
            i32x8 g1;
            g1[0] = (int)(1u << 16);                           // data_size = 1 (2 bytes)
            g1[1] = (int)(1600u << 16);                        // tensor_dim0[15:0] (k)
            g1[2] = (int)(1024u << 16);                        // dim0 hi=0 | tensor_dim1 (n)
            g1[3] = (int)(32u << 16);                          // dim1 hi=0 | tile_dim0 = 32
            g1[4] = 128;                                       // tile_dim1 = 128, tile_dim2 = 0
            g1[5] = 1600;                                      // tensor_dim0_stride[31:0]
            g1[6] = 0;                                         // stride0 hi | dim1_stride lo
            g1[7] = 0;
            i32x4 z4 = {0, 0, 0, 0};
#if __has_include(<hip/amd_detail/amd_gfx1250_TDM.h>)
            i32x8 z8 = {0, 0, 0, 0, 0, 0, 0, 0};
            __builtin_amdgcn_tensor_load_to_lds(g0, g1, z4, z4, z8, 0);
#else
            __builtin_amdgcn_tensor_load_to_lds(g0, g1, z4, z4, 0);
#endif
        }

        // ---- all waves: build A tile (pair features) while the DMA runs ----
        for (int e = 0; e < 16; ++e) {
            int idx = e * 256 + t;                     // 4096 elements
            int r = idx >> 5, kk = idx & 31;
            int m = mBase + r;
            int i = m / 50;
            int K = kB + kk;
            float v;
            if (K < 512)        v = g[(size_t)i * 512 + K];
            else if (K < 1024)  { int j = bestI[m]; v = g[(size_t)j * 512 + (K - 512)]; }
            else if (K < 1536)  { int j = bestI[m];
                                  v = g[(size_t)i * 512 + (K - 1024)] * g[(size_t)j * 512 + (K - 1024)]; }
            else                v = (float)phi[(size_t)m * 64 + (K - 1536)];
            As[r][kk] = (_Float16)v;
        }

        if (wave == 0) __builtin_amdgcn_s_wait_tensorcnt(0);
        __syncthreads();

        // ---- fragments + WMMA ----
        v16h bfrag[2];
        for (int sn = 0; sn < 2; ++sn) {
            int nb = wn * 32 + sn * 16 + l15;
            h8 lo = *(const h8*)&Bt[nb][hi * 8];
            h8 hh = *(const h8*)&Bt[nb][16 + hi * 8];
            bfrag[sn] = __builtin_shufflevector(lo, hh, 0,1,2,3,4,5,6,7,8,9,10,11,12,13,14,15);
        }
        for (int sm = 0; sm < 4; ++sm) {
            int mb = wm * 64 + sm * 16 + l15;
            h8 lo = *(const h8*)&As[mb][hi * 8];
            h8 hh = *(const h8*)&As[mb][16 + hi * 8];
            v16h afrag = __builtin_shufflevector(lo, hh, 0,1,2,3,4,5,6,7,8,9,10,11,12,13,14,15);
            for (int sn = 0; sn < 2; ++sn) {
                acc[sm][sn] = __builtin_amdgcn_wmma_f32_16x16x32_f16(
                    false, afrag, false, bfrag[sn], (short)0, acc[sm][sn], false, false);
            }
        }
        __syncthreads();
    }

    // ---- fused relu + dot(W2) epilogue; half-wave reduce; atomic accumulate ----
    for (int sm = 0; sm < 4; ++sm) {
        for (int sn = 0; sn < 2; ++sn) {
            int nb = nBase + wn * 32 + sn * 16 + l15;
            float w2 = W2p[nb];
            float bb = b1p[nb];
            for (int r = 0; r < 8; ++r) {
                float h = acc[sm][sn][r] + bb;
                h = h > 0.0f ? h : 0.0f;
                float p = h * w2;
                for (int off = 1; off < 16; off <<= 1)
                    p += __shfl_xor(p, off, 32);
                if (l15 == 0) {
                    int m = mBase + wm * 64 + sm * 16 + r + hi * 8;
                    atomicAdd(&s_out[m], p);
                }
            }
        }
    }
}

// ---------------------------------------------------------------------------
// Softmax over {eps, antecedents} + expected antecedent a_n  (one WG / row)
// ---------------------------------------------------------------------------
__global__ void softmax_an_kernel(const float* __restrict__ s, const float* __restrict__ bestS,
                                  const int* __restrict__ bestI, const float* __restrict__ g,
                                  const float* __restrict__ b2, float* __restrict__ a_n) {
    __shared__ float cv[51];
    __shared__ float p[51];
    __shared__ int   jj[51];
    int i = blockIdx.x;
    int t = threadIdx.x;                               // 128
    int nv = (i < 50) ? i : 50;
    if (t == 0) cv[0] = 0.0f;
    if (t >= 1 && t < 51) {
        int a = t - 1;
        cv[t] = (a < nv) ? (s[i * 50 + a] + b2[0] + bestS[i * 50 + a]) : NEGF;
        jj[t] = bestI[i * 50 + a];
    }
    __syncthreads();
    if (t == 0) {
        float mx = cv[0];
        for (int q = 1; q < 51; ++q) mx = fmaxf(mx, cv[q]);
        float sm = 0.0f;
        for (int q = 0; q < 51; ++q) { p[q] = __expf(cv[q] - mx); sm += p[q]; }
        float inv = 1.0f / sm;
        for (int q = 0; q < 51; ++q) p[q] *= inv;
    }
    __syncthreads();
    for (int c = t; c < 512; c += 128) {
        float acc = p[0] * g[(size_t)i * 512 + c];
        for (int a = 0; a < 50; ++a)
            acc += p[a + 1] * g[(size_t)jj[a + 1] * 512 + c];
        a_n[(size_t)i * 512 + c] = acc;
    }
}

// ---------------------------------------------------------------------------
// Gate GEMM: f_lin[1024,512] = [g | a_n] @ Wf   (fp32 WMMA 16x16x4)
// ---------------------------------------------------------------------------
__global__ void gate_gemm_kernel(const float* __restrict__ g, const float* __restrict__ a_n,
                                 const float* __restrict__ Wf, float* __restrict__ f_lin) {
    int mB = blockIdx.x * 16, nB = blockIdx.y * 16;
    int lane = threadIdx.x;
    int l15 = lane & 15, hi = lane >> 4;
    int m = mB + l15;
    v8f acc = {};
    for (int kb = 0; kb < 1024; kb += 4) {
        int k0 = kb + (hi ? 2 : 0);
        int k1 = k0 + 1;
        v2f a, b;
        a.x = (k0 < 512) ? g[(size_t)m * 512 + k0] : a_n[(size_t)m * 512 + (k0 - 512)];
        a.y = (k1 < 512) ? g[(size_t)m * 512 + k1] : a_n[(size_t)m * 512 + (k1 - 512)];
        b.x = Wf[(size_t)k0 * 512 + nB + l15];
        b.y = Wf[(size_t)k1 * 512 + nB + l15];
        acc = __builtin_amdgcn_wmma_f32_16x16x4_f32(false, a, false, b, (short)0, acc, false, false);
    }
    for (int r = 0; r < 8; ++r)
        f_lin[(size_t)(mB + r + hi * 8) * 512 + nB + l15] = acc[r];
}

// ---------------------------------------------------------------------------
// Gated refinement: g1 = f*g + (1-f)*a_n  (row 0 copied verbatim)
// ---------------------------------------------------------------------------
__global__ void refine_kernel(const float* __restrict__ g, const float* __restrict__ a_n,
                              const float* __restrict__ f_lin, const float* __restrict__ bf,
                              float* __restrict__ g1) {
    int idx = blockIdx.x * 256 + threadIdx.x;          // 1024*512
    if (idx >= 1024 * 512) return;
    int i = idx >> 9, c = idx & 511;
    float gi = g[idx];
    if (i == 0) { g1[idx] = gi; return; }
    float f = 1.0f / (1.0f + __expf(-(f_lin[idx] + bf[c])));
    g1[idx] = f * gi + (1.0f - f) * a_n[idx];
}

// ---------------------------------------------------------------------------
// Final score assembly: out [1024, 51]
// ---------------------------------------------------------------------------
__global__ void assemble_kernel(const float* __restrict__ s, const float* __restrict__ bestS,
                                const float* __restrict__ b2, float* __restrict__ out) {
    int idx = blockIdx.x * 256 + threadIdx.x;          // 1024*51
    if (idx >= 1024 * 51) return;
    int i = idx / 51, c = idx % 51;
    float v;
    if (c == 0)      v = 0.0f;                         // epsilon column
    else if (i == 0) v = (c == 1) ? 0.0f : NEGF;       // special row 0
    else {
        int a = c - 1;
        int nv = (i < 50) ? i : 50;
        v = (a < nv) ? (s[i * 50 + a] + b2[0] + bestS[i * 50 + a]) : NEGF;
    }
    out[idx] = v;
}

// ---------------------------------------------------------------------------
extern "C" void kernel_launch(void* const* d_in, const int* in_sizes, int n_in,
                              void* d_out, int out_size, void* d_ws, size_t ws_size,
                              hipStream_t stream) {
    const float* g      = (const float*)d_in[0];
    const float* ms     = (const float*)d_in[1];
    const float* distE  = (const float*)d_in[2];
    const float* genreE = (const float*)d_in[3];
    const float* spkE   = (const float*)d_in[4];
    const float* Wc     = (const float*)d_in[5];
    const float* W1     = (const float*)d_in[6];
    const float* b1     = (const float*)d_in[7];
    const float* W2     = (const float*)d_in[8];
    const float* b2     = (const float*)d_in[9];
    const float* Wf     = (const float*)d_in[10];
    const float* bf     = (const float*)d_in[11];
    const int* startI   = (const int*)d_in[12];
    const int* endI     = (const int*)d_in[13];
    const int* genreI   = (const int*)d_in[14];
    const int* spkI     = (const int*)d_in[15];
    float* out = (float*)d_out;

    char* ws = (char*)d_ws;
    auto carve = [&](size_t bytes) -> char* {
        char* p = ws;
        ws += (bytes + 255) & ~(size_t)255;
        return p;
    };
    float*     T     = (float*)    carve((size_t)1024 * 512 * 4);
    float*     ant   = (float*)    carve((size_t)1024 * 1024 * 4);
    float*     bestS = (float*)    carve((size_t)1024 * 50 * 4);
    int*       bestI = (int*)      carve((size_t)1024 * 50 * 4);
    _Float16*  phi   = (_Float16*) carve((size_t)51200 * 64 * 2);
    _Float16*  W1t   = (_Float16*) carve((size_t)1024 * 1600 * 2);
    float*     b1p   = (float*)    carve((size_t)1024 * 4);
    float*     W2p   = (float*)    carve((size_t)1024 * 4);
    float*     s0    = (float*)    carve((size_t)51200 * 4);
    float*     an    = (float*)    carve((size_t)1024 * 512 * 4);
    float*     flin  = (float*)    carve((size_t)1024 * 512 * 4);
    float*     g1    = (float*)    carve((size_t)1024 * 512 * 4);
    float*     s1    = (float*)    carve((size_t)51200 * 4);

    cvt_w1_kernel<<<(1024 * 1600 + 255) / 256, 256, 0, stream>>>(W1, W1t);
    cvt_aux_kernel<<<4, 256, 0, stream>>>(b1, W2, b1p, W2p);

    coarse1_kernel<<<dim3(64, 32), 32, 0, stream>>>(g, Wc, T);
    coarse2_kernel<<<dim3(64, 64), 32, 0, stream>>>(T, g, ms, ant);
    topk_kernel<<<1024, 256, 0, stream>>>(ant, bestS, bestI);
    phi_kernel<<<(51200 * 64 + 255) / 256, 256, 0, stream>>>(
        startI, endI, genreI, spkI, distE, genreE, spkE, bestI, phi);

    // iteration 0
    hipMemsetAsync(s0, 0, (size_t)51200 * 4, stream);
    mlp_gemm_kernel<<<dim3(400, 8), 256, 0, stream>>>(g, phi, bestI, W1t, b1p, W2p, s0);
    softmax_an_kernel<<<1024, 128, 0, stream>>>(s0, bestS, bestI, g, b2, an);
    gate_gemm_kernel<<<dim3(64, 32), 32, 0, stream>>>(g, an, Wf, flin);
    refine_kernel<<<(1024 * 512 + 255) / 256, 256, 0, stream>>>(g, an, flin, bf, g1);

    // iteration 1 (final)
    hipMemsetAsync(s1, 0, (size_t)51200 * 4, stream);
    mlp_gemm_kernel<<<dim3(400, 8), 256, 0, stream>>>(g1, phi, bestI, W1t, b1p, W2p, s1);

    assemble_kernel<<<(1024 * 51 + 255) / 256, 256, 0, stream>>>(s1, bestS, b2, out);
}
